// W2Attention_79078937854235
// MI455X (gfx1250) — matrix operand order repaired and checked by reading, not compiled, not run on hardware
//
#include <hip/hip_runtime.h>

// ---------------------------------------------------------------------------
// W2 (Wasserstein) attention for MI455X / gfx1250, wave32, WMMA f16->f32.
//
// Sizes: B=1, S=512, HID=512, NH=8, HD=64.
// - GEMMs (projections, P@V, output proj): v_wmma_f32_16x16x32_f16,
//   one 16x16 tile per wave, K step 32, b128 operand loads + WGP-scope
//   prefetch (locality=3 -> near scope; locality=0 gave useless SCOPE_SYS).
// - Pairwise score kernel is TRANS-bound: 1 v_rcp_f32 per element (cannot be
//   factored out), logs chunked 16-wide via products (16x fewer v_log_f32).
//   K tiles staged in LDS with GLOBAL_LOAD_ASYNC_TO_LDS_B128 (+ asynccnt).
// - Whole working set ~18 MB -> L2-resident (192 MB), compute-bound.
// ---------------------------------------------------------------------------

typedef __attribute__((ext_vector_type(16))) _Float16 v16h;
typedef __attribute__((ext_vector_type(8)))  _Float16 v8h;
typedef __attribute__((ext_vector_type(8)))  float    v8f;
typedef __attribute__((ext_vector_type(4)))  int      v4i;

constexpr int SEQ   = 512;
constexpr int HIDN  = 512;
constexpr int NHEAD = 8;
constexpr int HDIM  = 64;

// Async LDS path (gfx1250 GLOBAL_LOAD_ASYNC_TO_LDS_*). Guarded so the file
// still compiles (plain-load fallback) if the toolchain lacks the builtins.
// Probe (round 2 stderr) showed the builtin signature expects
// (int4 addrspace(1)* src, int4 addrspace(3)* dst, imm offset, imm cpol).
#if defined(__gfx1250__) && \
    __has_builtin(__builtin_amdgcn_global_load_async_to_lds_b128) && \
    __has_builtin(__builtin_amdgcn_s_wait_asynccnt)
#define W2_ASYNC_LDS 1
typedef __attribute__((address_space(1))) v4i* as1_v4i;   // global int4*
typedef __attribute__((address_space(3))) v4i* as3_v4i;   // LDS int4*
#endif

// --------------------------------------------------------------------------
// f32 -> f16 convert
// --------------------------------------------------------------------------
__global__ void f32_to_f16_kernel(const float* __restrict__ in,
                                  _Float16* __restrict__ out, int n) {
  int i = blockIdx.x * blockDim.x + threadIdx.x;
  if (i < n) out[i] = (_Float16)in[i];
}

// --------------------------------------------------------------------------
// C[M,N] (f32) = A[M,K] (f16) @ W[N,K]^T (f16)      ("tn": B is row-major NxK)
// block = 128 threads = 4 waves; wave -> one 16x16 tile; grid = (M/16, N/64)
//
// A fragment per lane (contiguous k order used by this code, same as B):
//   lanes 0-15 : K = kb+0..15   (aoff = 0)
//   lanes 16-31: K = kb+16..31  (aoff = 16)
// B fragment per lane (same contiguous k order as A): column n0+(lane&15),
//   lanes 0-15 : K = kb+0..15  (koff = 0)
//   lanes 16-31: K = kb+16..31 (koff = 16)
// C/D: VGPR r holds row m0+r (lanes 0-15) / m0+8+r (lanes 16-31), col n0+lane%16
// --------------------------------------------------------------------------
__global__ __launch_bounds__(128) void gemm_tn_wmma(
    const _Float16* __restrict__ A, const _Float16* __restrict__ W,
    float* __restrict__ C, int M, int N, int K) {
  const int lane  = threadIdx.x & 31;
  const int wave  = threadIdx.x >> 5;
  const int m0    = blockIdx.x * 16;
  const int n0    = blockIdx.y * 64 + wave * 16;
  const int hi    = lane >> 4;          // 0 or 1
  const int lr    = lane & 15;
  const int aoff  = hi * 16;
  const int koff  = hi * 16;
  const _Float16* arow = A + (size_t)(m0 + lr) * K;
  const _Float16* wrow = W + (size_t)(n0 + lr) * K;

  v8f c = {};
  for (int kb = 0; kb < K; kb += 32) {
    // near-scope prefetch: pull upcoming L2-resident lines into the WGP cache
    __builtin_prefetch(arow + kb + 256, 0, 3);
    __builtin_prefetch(wrow + kb + 256, 0, 3);
    v8h alo = *(const v8h*)(arow + kb + aoff);
    v8h ahi = *(const v8h*)(arow + kb + 8 + aoff);
    v16h a, b;
#pragma unroll
    for (int t = 0; t < 8; ++t) { a[t] = alo[t]; a[8 + t] = ahi[t]; }
    b = *(const v16h*)(wrow + kb + koff);
    c = __builtin_amdgcn_wmma_f32_16x16x32_f16(false, a, false, b,
                                               (short)0, c, false, false);
  }
#pragma unroll
  for (int r = 0; r < 8; ++r)
    C[(size_t)(m0 + r + hi * 8) * N + n0 + lr] = c[r];
}

// --------------------------------------------------------------------------
// Per-head:  C16[i, h*64+d] = sum_j A[h][i][j] * B[j, h*64+d]   ("nn")
// A = probs f16 [NHEAD][S][S], B = V f16 [S][HIDN], C f16 [S][HIDN]
// grid = (M/16, 1, NHEAD); block = 128 (4 waves cover the head's 64 cols)
// --------------------------------------------------------------------------
__global__ __launch_bounds__(128) void gemm_nn_wmma(
    const _Float16* __restrict__ A, const _Float16* __restrict__ B,
    _Float16* __restrict__ C, int M, int K, int ldb, int ldc) {
  const int h    = blockIdx.z;
  const int lane = threadIdx.x & 31;
  const int wave = threadIdx.x >> 5;
  const int m0   = blockIdx.x * 16;
  const int n0   = h * 64 + wave * 16;
  const int hi   = lane >> 4;
  const int lr   = lane & 15;
  const int aoff = hi * 16;
  const int koff = hi * 16;
  const _Float16* arow = A + (size_t)h * M * K + (size_t)(m0 + lr) * K;

  v8f c = {};
  for (int kb = 0; kb < K; kb += 32) {
    __builtin_prefetch(arow + kb + 256, 0, 3);
    v8h alo = *(const v8h*)(arow + kb + aoff);
    v8h ahi = *(const v8h*)(arow + kb + 8 + aoff);
    v16h a, b;
#pragma unroll
    for (int t = 0; t < 8; ++t) { a[t] = alo[t]; a[8 + t] = ahi[t]; }
    const _Float16* bp = B + (size_t)(kb + koff) * ldb + n0 + lr;
#pragma unroll
    for (int t = 0; t < 16; ++t) b[t] = bp[(size_t)t * ldb];
    c = __builtin_amdgcn_wmma_f32_16x16x32_f16(false, a, false, b,
                                               (short)0, c, false, false);
  }
#pragma unroll
  for (int r = 0; r < 8; ++r)
    C[(size_t)(m0 + r + hi * 8) * ldc + n0 + lr] = (_Float16)c[r];
}

// --------------------------------------------------------------------------
// Softplus + RoPE prep.
// Input proj: [S][2*HID] with per-head columns h*128 + (0..63 mu | 64..127 sig)
// Output: mu,sig as [h][s][d]  (idx = h*S*HD + s*HD + d)
// --------------------------------------------------------------------------
__global__ void rope_softplus_prep(const float* __restrict__ proj,
                                   const float* __restrict__ cosb,
                                   const float* __restrict__ sinb,
                                   float* __restrict__ mu,
                                   float* __restrict__ sig) {
  int idx = blockIdx.x * blockDim.x + threadIdx.x;
  if (idx >= NHEAD * SEQ * HDIM) return;
  int d = idx & (HDIM - 1);
  int s = (idx >> 6) & (SEQ - 1);
  int h = idx >> 15;
  const float* row = proj + (size_t)s * (2 * HIDN) + h * (2 * HDIM);
  float m    = row[d];
  float part = (d < HDIM / 2) ? -row[d + HDIM / 2] : row[d - HDIM / 2];
  float mr   = m * cosb[s * HDIM + d] + part * sinb[s * HDIM + d];
  float x    = row[HDIM + d];
  // softplus(x) = max(x,0) + log1p(exp(-|x|))
  float sp = fmaxf(x, 0.0f) + log1pf(__expf(-fabsf(x)));
  mu[idx]  = mr;
  sig[idx] = sp + 1e-4f;
}

// --------------------------------------------------------------------------
// Scores + softmax, f16 probs out.
// grid = (NHEAD, S/8); block = 256 = 8 waves; wave w -> query row i.
// K rows staged in LDS in 8 tiles of 64 rows via async-to-LDS b128 copies.
// Row stride 68 floats (272 B): 16B-aligned for B128, banks (4*jj+d)%64 ->
// worst case 2-way conflict (irrelevant: loop is TRANS-bound).
// log-det uses 16-wide products => 4 v_log_f32 per (i,j) instead of 64.
// Each lane holds 16 of the row's 512 scores (j = t*64 + c*32 + lane).
// --------------------------------------------------------------------------
__global__ __launch_bounds__(256) void scores_softmax(
    const float* __restrict__ qmu, const float* __restrict__ qsig,
    const float* __restrict__ kmu, const float* __restrict__ ksig,
    const float* __restrict__ mask, _Float16* __restrict__ probs) {
  constexpr int KSTR = 68;                      // LDS row stride in floats
  __shared__ __align__(16) float kmu_s[64 * KSTR];
  __shared__ __align__(16) float ksig_s[64 * KSTR];
  __shared__ float qmu_s[8][HDIM];
  __shared__ float qsig_s[8][HDIM];

  const int h    = blockIdx.x;
  const int wave = threadIdx.x >> 5;
  const int lane = threadIdx.x & 31;
  const int i    = blockIdx.y * 8 + wave;
  const size_t hb = (size_t)h * SEQ * HDIM;

  for (int d = lane; d < HDIM; d += 32) {
    qmu_s[wave][d]  = qmu[hb + (size_t)i * HDIM + d];
    qsig_s[wave][d] = qsig[hb + (size_t)i * HDIM + d];
  }

  float sc[16];
  for (int t = 0; t < 8; ++t) {
    __syncthreads();
#ifdef W2_ASYNC_LDS
    {
      const float* gk = kmu  + hb + (size_t)(t * 64) * HDIM;
      const float* gs = ksig + hb + (size_t)(t * 64) * HDIM;
      // tile = 64 rows x 64 floats = 1024 x 16B chunks per tensor
#pragma unroll
      for (int idx = threadIdx.x; idx < 1024; idx += 256) {
        int r = idx >> 4, c4 = (idx & 15) * 4;          // row, float col
        __builtin_amdgcn_global_load_async_to_lds_b128(
            (as1_v4i)(gk + r * HDIM + c4),
            (as3_v4i)&kmu_s[r * KSTR + c4], 0, 0);
        __builtin_amdgcn_global_load_async_to_lds_b128(
            (as1_v4i)(gs + r * HDIM + c4),
            (as3_v4i)&ksig_s[r * KSTR + c4], 0, 0);
      }
      __builtin_amdgcn_s_wait_asynccnt(0);
    }
#else
    for (int idx = threadIdx.x; idx < 64 * HDIM; idx += 256) {
      int r = idx >> 6, d = idx & (HDIM - 1);
      kmu_s[r * KSTR + d]  = kmu[hb + (size_t)(t * 64 + r) * HDIM + d];
      ksig_s[r * KSTR + d] = ksig[hb + (size_t)(t * 64 + r) * HDIM + d];
    }
#endif
    __syncthreads();
#pragma unroll
    for (int c = 0; c < 2; ++c) {
      int jj = c * 32 + lane;
      float wd = 0.0f, ld2 = 0.0f;      // ld2 accumulates log2(prod chunks)
#pragma unroll
      for (int dc = 0; dc < 4; ++dc) {  // 4 chunks of 16 dims
        float prod = 1.0f;
#pragma unroll
        for (int dd = 0; dd < 16; ++dd) {
          int d = dc * 16 + dd;
          float dm = qmu_s[wave][d] - kmu_s[jj * KSTR + d];
          float ss = qsig_s[wave][d] + ksig_s[jj * KSTR + d];
          wd   = __builtin_fmaf(dm * dm, __builtin_amdgcn_rcpf(ss), wd);
          prod *= ss;                   // Π ss stays in f32 range for 16 terms
        }
        ld2 += __builtin_amdgcn_logf(prod);   // v_log_f32 (log2)
      }
      int j = t * 64 + jj;
      sc[t * 2 + c] = -0.5f * (wd + ld2 * 0.69314718055994531f)
                      + mask[(size_t)i * SEQ + j];
    }
  }

  // softmax over row i (16 values/lane, wave32 butterfly reductions)
  float m = sc[0];
#pragma unroll
  for (int x = 1; x < 16; ++x) m = fmaxf(m, sc[x]);
#pragma unroll
  for (int off = 16; off >= 1; off >>= 1) m = fmaxf(m, __shfl_xor(m, off, 32));
  float tot = 0.0f;
#pragma unroll
  for (int x = 0; x < 16; ++x) {
    sc[x] = __builtin_amdgcn_exp2f((sc[x] - m) * 1.44269504088896340f);
    tot += sc[x];
  }
#pragma unroll
  for (int off = 16; off >= 1; off >>= 1) tot += __shfl_xor(tot, off, 32);
  float inv = __builtin_amdgcn_rcpf(tot);

  _Float16* prow = probs + (size_t)h * SEQ * SEQ + (size_t)i * SEQ;
#pragma unroll
  for (int x = 0; x < 16; ++x) {
    int t = x >> 1, c = x & 1;
    prow[t * 64 + c * 32 + lane] = (_Float16)(sc[x] * inv);
  }
}

// --------------------------------------------------------------------------
// Host-side launch
// --------------------------------------------------------------------------
extern "C" void kernel_launch(void* const* d_in, const int* in_sizes, int n_in,
                              void* d_out, int out_size, void* d_ws, size_t ws_size,
                              hipStream_t stream) {
  const float* hidden = (const float*)d_in[0];
  const float* cosb   = (const float*)d_in[1];
  const float* sinb   = (const float*)d_in[2];
  const float* mask   = (const float*)d_in[3];
  const float* wq     = (const float*)d_in[4];
  const float* wk     = (const float*)d_in[5];
  const float* wv     = (const float*)d_in[6];
  const float* wo     = (const float*)d_in[7];
  float* out          = (float*)d_out;

  // ---- carve scratch (total ~18 MB; assumes ws_size is sufficient) ----
  char* ws = (char*)d_ws;
  size_t off = 0;
  auto carve = [&](size_t bytes) -> void* {
    void* p = ws + off;
    off += (bytes + 255) & ~(size_t)255;
    return p;
  };
  _Float16* H16  = (_Float16*)carve((size_t)SEQ * HIDN * 2);        // hidden f16
  _Float16* WQ16 = (_Float16*)carve((size_t)2 * HIDN * HIDN * 2);
  _Float16* WK16 = (_Float16*)carve((size_t)2 * HIDN * HIDN * 2);
  _Float16* WV16 = (_Float16*)carve((size_t)HIDN * HIDN * 2);
  _Float16* WO16 = (_Float16*)carve((size_t)HIDN * HIDN * 2);
  float*    QF   = (float*)   carve((size_t)SEQ * 2 * HIDN * 4);    // q proj
  float*    KF   = (float*)   carve((size_t)SEQ * 2 * HIDN * 4);    // k proj
  float*    VF   = (float*)   carve((size_t)SEQ * HIDN * 4);        // v proj
  _Float16* V16  = (_Float16*)carve((size_t)SEQ * HIDN * 2);
  float*    QMU  = (float*)   carve((size_t)NHEAD * SEQ * HDIM * 4);
  float*    QSG  = (float*)   carve((size_t)NHEAD * SEQ * HDIM * 4);
  float*    KMU  = (float*)   carve((size_t)NHEAD * SEQ * HDIM * 4);
  float*    KSG  = (float*)   carve((size_t)NHEAD * SEQ * HDIM * 4);
  _Float16* P16  = (_Float16*)carve((size_t)NHEAD * SEQ * SEQ * 2); // probs
  _Float16* A16  = (_Float16*)carve((size_t)SEQ * HIDN * 2);        // attn out
  (void)ws_size; (void)n_in; (void)in_sizes; (void)out_size;

  auto cvt = [&](const float* src, _Float16* dst, int n) {
    f32_to_f16_kernel<<<(n + 255) / 256, 256, 0, stream>>>(src, dst, n);
  };

  // ---- 1) f16 copies of activations/weights ----
  cvt(hidden, H16,  SEQ * HIDN);
  cvt(wq,     WQ16, 2 * HIDN * HIDN);
  cvt(wk,     WK16, 2 * HIDN * HIDN);
  cvt(wv,     WV16, HIDN * HIDN);
  cvt(wo,     WO16, HIDN * HIDN);

  // ---- 2) projections (WMMA) ----
  gemm_tn_wmma<<<dim3(SEQ / 16, (2 * HIDN) / 64), 128, 0, stream>>>(
      H16, WQ16, QF, SEQ, 2 * HIDN, HIDN);
  gemm_tn_wmma<<<dim3(SEQ / 16, (2 * HIDN) / 64), 128, 0, stream>>>(
      H16, WK16, KF, SEQ, 2 * HIDN, HIDN);
  gemm_tn_wmma<<<dim3(SEQ / 16, HIDN / 64), 128, 0, stream>>>(
      H16, WV16, VF, SEQ, HIDN, HIDN);
  cvt(VF, V16, SEQ * HIDN);

  // ---- 3) softplus + RoPE prep ----
  {
    int n = NHEAD * SEQ * HDIM;
    rope_softplus_prep<<<(n + 255) / 256, 256, 0, stream>>>(QF, cosb, sinb, QMU, QSG);
    rope_softplus_prep<<<(n + 255) / 256, 256, 0, stream>>>(KF, cosb, sinb, KMU, KSG);
  }

  // ---- 4) pairwise W2 scores + softmax -> f16 probs ----
  scores_softmax<<<dim3(NHEAD, SEQ / 8), 256, 0, stream>>>(
      QMU, QSG, KMU, KSG, mask, P16);

  // ---- 5) P @ V per head (WMMA), f16 out in [s][h*64+d] layout ----
  gemm_nn_wmma<<<dim3(SEQ / 16, 1, NHEAD), 128, 0, stream>>>(
      P16, V16, A16, SEQ, SEQ, HIDN, HIDN);

  // ---- 6) output projection (WMMA) -> f32 d_out ----
  gemm_tn_wmma<<<dim3(SEQ / 16, HIDN / 64), 128, 0, stream>>>(
      A16, WO16, out, SEQ, HIDN, HIDN);
}
